// BitFeedForward_30631706755716
// MI455X (gfx1250) — 8-bit matrix operand order repaired and checked by reading, compiled, not run on hardware
//
#include <hip/hip_runtime.h>
#include <stdint.h>

typedef __attribute__((ext_vector_type(8))) int v8i;
typedef __attribute__((ext_vector_type(4))) int i4v;

#define EPSV 1e-5f

#if __has_builtin(__builtin_amdgcn_s_wait_asynccnt)
#define WAIT_ASYNC(n) __builtin_amdgcn_s_wait_asynccnt(n)
#else
#define WAIT_ASYNC(n) asm volatile("s_wait_asynccnt %0" ::"i"(n) : "memory")
#endif

__device__ __forceinline__ void async_cp16(unsigned lds_addr, const void* gptr) {
    asm volatile("global_load_async_to_lds_b128 %0, %1, off" ::"v"(lds_addr), "v"(gptr)
                 : "memory");
}

__global__ __launch_bounds__(256) void absmean_partial(const float* __restrict__ w,
                                                       long long n, float* __restrict__ part) {
    __shared__ float sm[256];
    float s = 0.f;
    for (long long i = (long long)blockIdx.x * 256 + threadIdx.x; i < n;
         i += (long long)gridDim.x * 256)
        s += fabsf(w[i]);
    sm[threadIdx.x] = s; __syncthreads();
    for (int o = 128; o > 0; o >>= 1) {
        if ((int)threadIdx.x < o) sm[threadIdx.x] += sm[threadIdx.x + o];
        __syncthreads();
    }
    if (threadIdx.x == 0) part[blockIdx.x] = sm[0];
}

__global__ __launch_bounds__(256) void absmean_final(const float* __restrict__ part, int np,
                                                     long long n, float* __restrict__ scale) {
    __shared__ float sm[256];
    float s = 0.f;
    for (int i = threadIdx.x; i < np; i += 256) s += part[i];
    sm[threadIdx.x] = s; __syncthreads();
    for (int o = 128; o > 0; o >>= 1) {
        if ((int)threadIdx.x < o) sm[threadIdx.x] += sm[threadIdx.x + o];
        __syncthreads();
    }
    if (threadIdx.x == 0) scale[0] = fmaxf(sm[0] / (float)n, EPSV);
}

__global__ __launch_bounds__(256) void ternarize(const float* __restrict__ w, long long n,
                                                 const float* __restrict__ scale,
                                                 signed char* __restrict__ w8) {
    const float inv = 1.0f / scale[0];
    for (long long i = (long long)blockIdx.x * blockDim.x + threadIdx.x; i < n;
         i += (long long)gridDim.x * blockDim.x) {
        float q = rintf(w[i] * inv);
        q = fminf(1.f, fmaxf(-1.f, q));
        w8[i] = (signed char)q;
    }
}

__global__ __launch_bounds__(256) void act_quant(const float* __restrict__ x, int H,
                                                 signed char* __restrict__ x8,
                                                 float* __restrict__ invS) {
    __shared__ float sm[256];
    const float* row = x + (size_t)blockIdx.x * H;
    float m = 0.f;
    for (int i = threadIdx.x; i < H; i += 256) m = fmaxf(m, fabsf(row[i]));
    sm[threadIdx.x] = m; __syncthreads();
    for (int o = 128; o > 0; o >>= 1) {
        if ((int)threadIdx.x < o) sm[threadIdx.x] = fmaxf(sm[threadIdx.x], sm[threadIdx.x + o]);
        __syncthreads();
    }
    const float mx = fmaxf(sm[0], EPSV);
    const float sc = 127.f / mx;
    signed char* orow = x8 + (size_t)blockIdx.x * H;
    for (int i = threadIdx.x; i < H; i += 256) {
        float q = rintf(row[i] * sc);
        q = fminf(127.f, fmaxf(-127.f, q));
        orow[i] = (signed char)q;
    }
    if (threadIdx.x == 0) invS[blockIdx.x] = mx / 127.f;
}

__global__ __launch_bounds__(256) void quant_h(const _Float16* __restrict__ h, int W,
                                               signed char* __restrict__ h8,
                                               float* __restrict__ invS) {
    __shared__ float sm[256];
    const _Float16* row = h + (size_t)blockIdx.x * W;
    float m = 0.f;
    for (int i = threadIdx.x; i < W; i += 256) m = fmaxf(m, (float)row[i]);
    sm[threadIdx.x] = m; __syncthreads();
    for (int o = 128; o > 0; o >>= 1) {
        if ((int)threadIdx.x < o) sm[threadIdx.x] = fmaxf(sm[threadIdx.x], sm[threadIdx.x + o]);
        __syncthreads();
    }
    const float mx = fmaxf(sm[0], EPSV);
    const float sc = 127.f / mx;
    signed char* orow = h8 + (size_t)blockIdx.x * W;
    for (int i = threadIdx.x; i < W; i += 256) {
        float q = rintf((float)row[i] * sc);
        q = fminf(127.f, q);
        orow[i] = (signed char)q;
    }
    if (threadIdx.x == 0) invS[blockIdx.x] = mx / 127.f;
}

union FragA { v8i v; long long d[4]; };
union FragB { v8i v; i4v q[2]; };

#define LDS_STRIDE 80
#define TILE_BYTES (128 * LDS_STRIDE)
#define BUF_BYTES  (2 * TILE_BYTES)

template <int EPI>
__global__ __launch_bounds__(256) void gemm_i8(const signed char* __restrict__ A,
                                               const signed char* __restrict__ Bw,
                                               int M, int N, int K,
                                               const float* __restrict__ wScale,
                                               const float* __restrict__ invRow,
                                               void* __restrict__ outP) {
    __shared__ signed char smem[2 * BUF_BYTES];

    const int tid   = threadIdx.x;
    const int wave  = tid >> 5;
    const int lane  = tid & 31;
    const int l     = lane & 15;
    const int hi    = lane >> 4;
    const int mBlk  = blockIdx.y * 128;
    const int nBlk  = blockIdx.x * 128;
    const int waveM = (wave & 1) * 64;
    const int waveN = (wave >> 1) * 32;

    const unsigned ldsBase = (unsigned)(uintptr_t)&smem[0];

    const int cRow = tid >> 2;
    const int cOff = (tid & 3) * 16;

    auto stage = [&](int k0, int buf) {
        const unsigned bufBase = ldsBase + buf * BUF_BYTES;
#pragma unroll
        for (int i = 0; i < 2; ++i) {
            const int row = cRow + i * 64;
            async_cp16(bufBase + row * LDS_STRIDE + cOff,
                       A + (size_t)(mBlk + row) * K + k0 + cOff);
        }
#pragma unroll
        for (int i = 0; i < 2; ++i) {
            const int row = cRow + i * 64;
            async_cp16(bufBase + TILE_BYTES + row * LDS_STRIDE + cOff,
                       Bw + (size_t)(nBlk + row) * K + k0 + cOff);
        }
    };

    const v8i vzero = {};
    v8i acc[4][2];
#pragma unroll
    for (int mi = 0; mi < 4; ++mi)
#pragma unroll
        for (int ni = 0; ni < 2; ++ni) acc[mi][ni] = vzero;

    auto compute = [&](int buf) {
        const signed char* sA = smem + buf * BUF_BYTES;
        const signed char* sB = sA + TILE_BYTES;

        FragA a[4];
#pragma unroll
        for (int mi = 0; mi < 4; ++mi) {
            const signed char* p = sA + (waveM + mi * 16 + l) * LDS_STRIDE + 16 * hi;
#pragma unroll
            for (int j = 0; j < 4; ++j)
                a[mi].d[j] = *(const long long*)(p + 8 * (j & 1) + 32 * (j >> 1));
        }
        FragB b[2];
#pragma unroll
        for (int ni = 0; ni < 2; ++ni) {
            const signed char* p = sB + (waveN + ni * 16 + l) * LDS_STRIDE + 16 * hi;
            b[ni].q[0] = *(const i4v*)(p);
            b[ni].q[1] = *(const i4v*)(p + 32);
        }
#pragma unroll
        for (int mi = 0; mi < 4; ++mi)
#pragma unroll
            for (int ni = 0; ni < 2; ++ni)
                acc[mi][ni] = __builtin_amdgcn_wmma_i32_16x16x64_iu8(
                     true, a[mi].v,  true, b[ni].v,
                    acc[mi][ni],  false,  false);
    };

    stage(0, 0);
    int buf = 0;
    int k0 = 0;
    for (; k0 + 64 < K; k0 += 64) {
        stage(k0 + 64, buf ^ 1);
        WAIT_ASYNC(4);
        __syncthreads();
        compute(buf);
        __syncthreads();
        buf ^= 1;
    }
    WAIT_ASYNC(0);
    __syncthreads();
    compute(buf);

    const float sw = wScale[0];
#pragma unroll
    for (int mi = 0; mi < 4; ++mi) {
#pragma unroll
        for (int r = 0; r < 8; ++r) {
            const int row = mBlk + waveM + mi * 16 + hi * 8 + r;
            const float dq = sw * invRow[row];
#pragma unroll
            for (int ni = 0; ni < 2; ++ni) {
                const int col = nBlk + waveN + ni * 16 + l;
                float f = (float)acc[mi][ni][r] * dq;
                if (EPI == 0) {
                    f = f > 0.f ? f * f : 0.f;
                    ((_Float16*)outP)[(size_t)row * N + col] = (_Float16)f;
                } else {
                    ((float*)outP)[(size_t)row * N + col] = f;
                }
            }
        }
    }
}

extern "C" void kernel_launch(void* const* d_in, const int* in_sizes, int n_in,
                              void* d_out, int out_size, void* d_ws, size_t ws_size,
                              hipStream_t stream) {
    const float* x      = (const float*)d_in[0];
    const float* w_up   = (const float*)d_in[1];
    const float* w_down = (const float*)d_in[2];

    const int H = 2048, I = 4096;
    const int M = 4 * 4096;
    const long long NU = (long long)I * H;
    const long long ND = (long long)H * I;

    char* ws = (char*)d_ws;
    size_t off = 0;
    auto alloc = [&](size_t bytes) -> void* {
        void* p = ws + off;
        off += (bytes + 255) & ~(size_t)255;
        return p;
    };
    float*       partU  = (float*)alloc(512 * 4);
    float*       partD  = (float*)alloc(512 * 4);
    float*       scaleU = (float*)alloc(4);
    float*       scaleD = (float*)alloc(4);
    float*       invSx  = (float*)alloc((size_t)M * 4);
    float*       invSh  = (float*)alloc((size_t)M * 4);
    signed char* x8     = (signed char*)alloc((size_t)M * H);
    signed char* wu8    = (signed char*)alloc((size_t)NU);
    signed char* wd8    = (signed char*)alloc((size_t)ND);
    _Float16*    hHalf  = (_Float16*)alloc((size_t)M * I * 2);
    signed char* h8     = (signed char*)alloc((size_t)M * I);

    absmean_partial<<<512, 256, 0, stream>>>(w_up, NU, partU);
    absmean_partial<<<512, 256, 0, stream>>>(w_down, ND, partD);
    absmean_final<<<1, 256, 0, stream>>>(partU, 512, NU, scaleU);
    absmean_final<<<1, 256, 0, stream>>>(partD, 512, ND, scaleD);
    ternarize<<<2048, 256, 0, stream>>>(w_up, NU, scaleU, wu8);
    ternarize<<<2048, 256, 0, stream>>>(w_down, ND, scaleD, wd8);
    act_quant<<<M, 256, 0, stream>>>(x, H, x8, invSx);

    dim3 g1(I / 128, M / 128);
    gemm_i8<0><<<g1, 256, 0, stream>>>(x8, wu8, M, I, H, scaleU, invSx, (void*)hHalf);

    quant_h<<<M, 256, 0, stream>>>(hHalf, I, h8, invSh);

    dim3 g2(H / 128, M / 128);
    gemm_i8<1><<<g2, 256, 0, stream>>>(h8, wd8, M, H, I, scaleD, invSh, d_out);
}
